// HexConv2d_79053168050679
// MI455X (gfx1250) — compile-verified, matrix operand order checked by reading, not run on hardware
//
#include <hip/hip_runtime.h>
#include <hip/hip_bf16.h>

// ---------------------------------------------------------------------------
// HexConv2d as implicit GEMM on gfx1250 WMMA.
//   out[b,o,y,x] = mask(y,x) * ( bias[o] + sum_{n=0..6} sum_{c} w[o,c,n] *
//                                (x*mask)[b,c,y+dy_n,x+dx_n] )
// M = Cout(256), N = spatial(32*49*49), K = Cin*7 (1792)
// f16 operands, f32 accumulation via v_wmma_f32_16x16x32_f16.
// Roofline: 70.5 GFLOP, ~240 MB HBM traffic -> memory-bound at f16 WMMA rates;
// all effort goes into data movement (NHWC f16 pack, LDS slab reuse x16,
// async global->LDS staging).
// ---------------------------------------------------------------------------

typedef __attribute__((ext_vector_type(16))) _Float16 v16h;
typedef __attribute__((ext_vector_type(8)))  _Float16 v8h;
typedef __attribute__((ext_vector_type(8)))  float    v8f;
typedef int v4i __attribute__((__vector_size__(16)));   // builtin's b128 unit
typedef __attribute__((address_space(1))) v4i* gptr_v4i;
typedef __attribute__((address_space(3))) v4i* lptr_v4i;

#define BB   32
#define CIN  256
#define COUT 256
#define HH   49
#define WW   49
#define HP   51   // padded H (1 halo each side)
#define WP   66   // padded W (1 left halo, tiles reach gx=65)

#if defined(__has_builtin)
#if __has_builtin(__builtin_amdgcn_global_load_async_to_lds_b128)
#define USE_ASYNC_LDS 1
#endif
#endif

// hex mask for H=49: |(y-24)+(x-24)| <= 24  <=>  |y+x-48| <= 24
__device__ __forceinline__ bool hex_mask(int y, int x) {
    int s = y + x - 48;
    return (s < 0 ? -s : s) <= 24;
}

// ---------------------------------------------------------------------------
// Pack weights [O,C,7] fp32 -> per-lane WMMA A-fragment layout, f16.
// Flat index: (((n*8 + cc)*16 + mb)*32 + lane)*16 + h
//   lane<16 : M = mb*16+lane,     halves h: K = cc*32 + {0..7, 16..23}
//   lane>=16: M = mb*16+lane-16,  halves h: K = cc*32 + {8..15, 24..31}
// ---------------------------------------------------------------------------
__global__ __launch_bounds__(256) void pack_w_kernel(
        const float* __restrict__ w, _Float16* __restrict__ wp) {
    int idx  = blockIdx.x * 256 + threadIdx.x;      // < 7*8*16*32*16 = 458752
    int h    = idx & 15;
    int lane = (idx >> 4) & 31;
    int mb   = (idx >> 9) & 15;
    int cc   = (idx >> 13) & 7;
    int n    = idx >> 16;
    int M    = mb * 16 + (lane & 15);
    int klo  = (lane >> 4) * 8;
    int kk   = (h < 8) ? (klo + h) : (klo + 8 + h); // K{klo..+7, klo+16..+23}
    int c    = cc * 32 + kk;
    wp[idx]  = (_Float16)w[(M * CIN + c) * 7 + n];
}

// ---------------------------------------------------------------------------
// Pack x [B,C,H,W] fp32 -> masked, haloed NHWC f16: xm[b][gy][gx][c].
// Transpose 32ch x 64col tiles through LDS so both sides stay coalesced.
// grid = (8 cblk, 49 y, 32 b), block = 256
// ---------------------------------------------------------------------------
__global__ __launch_bounds__(256) void pack_x_kernel(
        const float* __restrict__ x, _Float16* __restrict__ xm) {
    __shared__ _Float16 t[64 * 33];
    int cblk = blockIdx.x, y = blockIdx.y, b = blockIdx.z;
    int tid  = threadIdx.x;

    for (int idx = tid; idx < 32 * 64; idx += 256) {
        int c_local = idx >> 6, xc = idx & 63;
        float v = 0.f;
        if (xc < WW && hex_mask(y, xc))
            v = x[(((size_t)(b * CIN + cblk * 32 + c_local) * HH + y) * WW) + xc];
        t[xc * 33 + c_local] = (_Float16)v;
    }
    __syncthreads();
    for (int idx = tid; idx < WW * 32; idx += 256) {
        int xc = idx >> 5, c_local = idx & 31;
        xm[((size_t)(b * HP + (y + 1)) * WP + (xc + 1)) * CIN + cblk * 32 + c_local] =
            t[xc * 33 + c_local];
    }
}

// ---------------------------------------------------------------------------
// Main kernel. grid = B*H*4 blocks, 512 threads = 16 waves.
// Block owns (b, y, x0..x0+15); stages the 3x18x256 f16 slab into LDS once
// (async direct-to-LDS when available), 16 waves (one per 16-Cout block)
// each run 56 v_wmma_f32_16x16x32_f16.
// ---------------------------------------------------------------------------
__global__ __launch_bounds__(512) void hexconv_wmma_kernel(
        const _Float16* __restrict__ xm, const _Float16* __restrict__ wp,
        const float* __restrict__ bias, float* __restrict__ out) {
    __shared__ _Float16 sm[3 * 18 * CIN];           // 27,648 bytes

    int bidx = blockIdx.x;
    int xt = bidx & 3;
    int y  = (bidx >> 2) % HH;
    int b  = bidx / (4 * HH);
    int x0 = xt << 4;
    int tid = threadIdx.x;

    // ---- stage input slab: 3 rows, each a contiguous 18*256-half run ------
    // (row-wise flat chunks: no div/mod in the address math)
#pragma unroll
    for (int r = 0; r < 3; ++r) {
        const _Float16* gbase = xm + ((size_t)(b * HP + y + r) * WP + x0) * CIN;
        _Float16* lbase = &sm[r * 18 * CIN];
        for (int i = tid; i < 576; i += 512) {      // 576 chunks of 8 halves
            int off = i * 8;
#if defined(USE_ASYNC_LDS)
            __builtin_amdgcn_global_load_async_to_lds_b128(
                (gptr_v4i)(gbase + off), (lptr_v4i)(lbase + off), 0, 0);
#else
            *(v8h*)&lbase[off] = *(const v8h*)&gbase[off];
#endif
        }
    }
#if defined(USE_ASYNC_LDS)
#if __has_builtin(__builtin_amdgcn_s_wait_asynccnt)
    __builtin_amdgcn_s_wait_asynccnt(0);
#else
    asm volatile("s_wait_asynccnt 0" ::: "memory");
#endif
#endif
    __syncthreads();

    const int wv   = tid >> 5;                      // Cout block (16 per wave)
    const int lane = tid & 31;
    const int nn   = lane & 15;                     // N (spatial col in tile)
    const int klo  = (lane >> 4) << 3;              // K-half select per layout

    static const int DY[7] = { 0, 1, 0, -1, -1, 0, 1 };
    static const int DX[7] = { 0, 0, 1,  1,  0,-1,-1 };

    v8f acc0 = {}; v8f acc1 = {};                   // two chains for XDL ILP
    for (int n = 0; n < 7; ++n) {
        const int row = DY[n] + 1;
        const int xc  = nn + DX[n] + 1;
        const _Float16* bsrc = &sm[(row * 18 + xc) * CIN + klo];
        const _Float16* abase =
            &wp[(size_t)(((n * 8) * 16 + wv) * 32 + lane) * 16];
        if (n < 6)  // prefetch next direction's weight block (global_prefetch_b8)
            __builtin_prefetch(abase + 8 * 16 * 32 * 16, 0, 1);
#pragma unroll
        for (int cc = 0; cc < 8; ++cc) {
            const v16h a = *(const v16h*)&abase[(size_t)cc * 16 * 32 * 16];
            v8h b0 = *(const v8h*)&bsrc[cc * 32];
            v8h b1 = *(const v8h*)&bsrc[cc * 32 + 16];
            v16h bm = __builtin_shufflevector(b0, b1,
                       0,1,2,3,4,5,6,7,8,9,10,11,12,13,14,15);
            if (cc & 1)
                acc1 = __builtin_amdgcn_wmma_f32_16x16x32_f16(
                        false, a, false, bm, (short)0, acc1, false, false);
            else
                acc0 = __builtin_amdgcn_wmma_f32_16x16x32_f16(
                        false, a, false, bm, (short)0, acc0, false, false);
        }
    }

    // ---- store D: M = r + 8*(lane>=16), N = lane&15 (ISA 7.12.2 layout) ----
    const int xg = x0 + nn;
    if (xg < WW) {
        const bool mk = hex_mask(y, xg);
        const int Mbase = wv * 16 + ((lane >> 4) << 3);
#pragma unroll
        for (int r = 0; r < 8; ++r) {
            const int M = Mbase + r;
            const float v = mk ? (acc0[r] + acc1[r] + bias[M]) : 0.f;
            out[((size_t)(b * COUT + M) * HH + y) * WW + xg] = v;
        }
    }
}

// ---------------------------------------------------------------------------
extern "C" void kernel_launch(void* const* d_in, const int* in_sizes, int n_in,
                              void* d_out, int out_size, void* d_ws, size_t ws_size,
                              hipStream_t stream) {
    const float* x    = (const float*)d_in[0];   // [32,256,49,49]
    const float* w    = (const float*)d_in[1];   // [256,256,7]
    const float* bias = (const float*)d_in[2];   // [256]
    float* out = (float*)d_out;                  // [32,256,49,49]

    _Float16* wp = (_Float16*)d_ws;                              // 896 KB
    _Float16* xm = (_Float16*)((char*)d_ws + (1u << 20));        // 55.1 MB
    const size_t xm_bytes = (size_t)BB * HP * WP * CIN * sizeof(_Float16);

    (void)hipMemsetAsync(xm, 0, xm_bytes, stream);   // zero halo (capturable)
    pack_w_kernel<<<458752 / 256, 256, 0, stream>>>(w, wp);
    pack_x_kernel<<<dim3(8, HH, BB), 256, 0, stream>>>(x, xm);
    hexconv_wmma_kernel<<<BB * HH * 4, 512, 0, stream>>>(xm, wp, bias, out);
    (void)in_sizes; (void)n_in; (void)out_size; (void)ws_size;
}
